// MultiHeadedAttention_15255723836019
// MI455X (gfx1250) — matrix operand order repaired and checked by reading, not compiled, not run on hardware
//
#include <hip/hip_runtime.h>
#include <hip/hip_bf16.h>

typedef _Float16 half_t;
typedef __attribute__((ext_vector_type(4)))  _Float16 v4h;
typedef __attribute__((ext_vector_type(8)))  _Float16 v8h;
typedef __attribute__((ext_vector_type(16))) _Float16 v16h;
typedef __attribute__((ext_vector_type(8)))  float    v8f;

#define B_  4
#define S_  2048
#define D_  1024
#define H_  16
#define DK_ 64
#define M_  (B_ * S_)   // 8192 rows for the projection GEMMs

// ---------------------------------------------------------------- fragments
__device__ __forceinline__ v16h cat16(v8h lo, v8h hi) {
  v16h r;
#pragma unroll
  for (int i = 0; i < 8; ++i) { r[i] = lo[i]; r[i + 8] = hi[i]; }
  return r;
}

// A fragment 16x32 f16 (MxK). Lane<16: K 0-7 & 16-23; lane>=16: K 8-15 & 24-31.
__device__ __forceinline__ v16h load_a_frag(const half_t* p, int ld, int lane) {
  int row = lane & 15;
  int k0  = (lane >> 4) << 3;        // 0 or 8
  const half_t* q = p + row * ld;
  return cat16(*(const v8h*)(q + k0), *(const v8h*)(q + 16 + k0));
}

// B fragment 32x16 f16 (KxN) from source laid out row-major [n][k]
// Lane<16: K 0-7 & 16-23; lane>=16: K 8-15 & 24-31; n = lane&15.
__device__ __forceinline__ v16h load_b_frag(const half_t* p, int ld, int lane) {
  int n  = lane & 15;
  int kb = (lane >> 4) << 3;          // 0 or 8
  const half_t* q = p + n * ld + kb;
  return cat16(*(const v8h*)(q), *(const v8h*)(q + 16));
}

__device__ __forceinline__ v8f wmma_f16(v16h a, v16h b, v8f c) {
  return __builtin_amdgcn_wmma_f32_16x16x32_f16(false, a, false, b,
                                                (short)0, c, false, false);
}

// -------------------------------------------- async global->LDS (ASYNCcnt)
// One instruction moves 16B per lane from global to LDS, no VGPR staging.
__device__ __forceinline__ void async_b128(unsigned lds_off, const half_t* g) {
  unsigned long long ga = (unsigned long long)(size_t)g;
  asm volatile("global_load_async_to_lds_b128 %0, %1, off"
               :: "v"(lds_off), "v"(ga) : "memory");
}
__device__ __forceinline__ void wait_async_le8() {
  asm volatile("s_wait_asynccnt 0x8" ::: "memory");
}
__device__ __forceinline__ void wait_async_0() {
  asm volatile("s_wait_asynccnt 0x0" ::: "memory");
}

// ---------------------------------------------------------------- fp32->f16
__global__ __launch_bounds__(256) void cvt_f32_f16(const float* __restrict__ in,
                                                   half_t* __restrict__ out,
                                                   int n) {
  int i = (blockIdx.x * blockDim.x + threadIdx.x) << 2;
  if (i + 3 < n) {
    float4 f = *(const float4*)(in + i);
    v4h h = {(half_t)f.x, (half_t)f.y, (half_t)f.z, (half_t)f.w};
    *(v4h*)(out + i) = h;
  }
}

// ------------------------------------------------- QKV projection GEMMs
// C(8192x1024) = X(f16, MxK) @ W(f16, NxK)^T + bias, scattered into the
// reference's reshape(B,H,S,DK) head layout, stored f16.
// 128x128 tile, BK=64, double-buffered async global->LDS pipeline.
__global__ __launch_bounds__(256) void gemm_qkv_kernel(
    const half_t* __restrict__ X, const half_t* __restrict__ W,
    const float* __restrict__ bias, half_t* __restrict__ OutH) {
  __shared__ __align__(16) half_t As[2][128][72];
  __shared__ __align__(16) half_t Bs[2][128][72];
  int tid = threadIdx.x, lane = tid & 31, w = tid >> 5;
  int mb = blockIdx.y * 128, nb = blockIdx.x * 128;
  int wr = (w & 3) * 32, wc = (w >> 2) * 64;   // wave sub-tile 32x64
  unsigned ldsA[2] = {(unsigned)(size_t)&As[0][0][0], (unsigned)(size_t)&As[1][0][0]};
  unsigned ldsB[2] = {(unsigned)(size_t)&Bs[0][0][0], (unsigned)(size_t)&Bs[1][0][0]};

  auto issue = [&](int buf, int kk) {              // 8 async insts / wave
#pragma unroll
    for (int i = 0; i < 4; ++i) {
      int slot = tid + 256 * i;                    // 1024 chunks of 16B
      int r = slot >> 3, c = (slot & 7) << 3;
      unsigned off = (unsigned)((r * 72 + c) * 2);
      async_b128(ldsA[buf] + off, X + (size_t)(mb + r) * D_ + kk + c);
      async_b128(ldsB[buf] + off, W + (size_t)(nb + r) * D_ + kk + c);
    }
  };

  v8f acc[2][4] = {};
  issue(0, 0);
  for (int kk = 0; kk < D_; kk += 64) {
    int buf = (kk >> 6) & 1;
    bool more = (kk + 64) < D_;
    if (more) { issue(buf ^ 1, kk + 64); wait_async_le8(); }
    else      { wait_async_0(); }
    __syncthreads();
#pragma unroll
    for (int k2 = 0; k2 < 64; k2 += 32) {
      v16h af[2], bf[4];
#pragma unroll
      for (int i = 0; i < 2; ++i) af[i] = load_a_frag(&As[buf][wr + 16 * i][k2], 72, lane);
#pragma unroll
      for (int j = 0; j < 4; ++j) bf[j] = load_b_frag(&Bs[buf][wc + 16 * j][k2], 72, lane);
#pragma unroll
      for (int i = 0; i < 2; ++i)
#pragma unroll
        for (int j = 0; j < 4; ++j) acc[i][j] = wmma_f16(af[i], bf[j], acc[i][j]);
    }
    __syncthreads();
  }
  // bias + scatter into head layout: h=r/128, s=(r%128)*16+c/64, dk=c%64
#pragma unroll
  for (int i = 0; i < 2; ++i)
#pragma unroll
    for (int j = 0; j < 4; ++j) {
      int ncol  = nb + wc + 16 * j + (lane & 15);
      float bv  = bias[ncol];
      int mbase = mb + wr + 16 * i + ((lane & 16) ? 8 : 0);
#pragma unroll
      for (int r = 0; r < 8; ++r) {
        int m = mbase + r;
        float val = acc[i][j][r] + bv;
        int b  = m >> 11, rr = m & 2047;
        int h  = rr >> 7;
        int s  = ((rr & 127) << 4) | (ncol >> 6);
        int dk = ncol & 63;
        OutH[(((size_t)(b * H_ + h)) * S_ + s) * DK_ + dk] = (half_t)val;
      }
    }
}

// ------------------------------------------------- output projection GEMM
__global__ __launch_bounds__(256) void gemm_out_kernel(
    const half_t* __restrict__ X, const half_t* __restrict__ W,
    const float* __restrict__ bias, float* __restrict__ Out) {
  __shared__ __align__(16) half_t As[2][128][72];
  __shared__ __align__(16) half_t Bs[2][128][72];
  int tid = threadIdx.x, lane = tid & 31, w = tid >> 5;
  int mb = blockIdx.y * 128, nb = blockIdx.x * 128;
  int wr = (w & 3) * 32, wc = (w >> 2) * 64;
  unsigned ldsA[2] = {(unsigned)(size_t)&As[0][0][0], (unsigned)(size_t)&As[1][0][0]};
  unsigned ldsB[2] = {(unsigned)(size_t)&Bs[0][0][0], (unsigned)(size_t)&Bs[1][0][0]};

  auto issue = [&](int buf, int kk) {
#pragma unroll
    for (int i = 0; i < 4; ++i) {
      int slot = tid + 256 * i;
      int r = slot >> 3, c = (slot & 7) << 3;
      unsigned off = (unsigned)((r * 72 + c) * 2);
      async_b128(ldsA[buf] + off, X + (size_t)(mb + r) * D_ + kk + c);
      async_b128(ldsB[buf] + off, W + (size_t)(nb + r) * D_ + kk + c);
    }
  };

  v8f acc[2][4] = {};
  issue(0, 0);
  for (int kk = 0; kk < D_; kk += 64) {
    int buf = (kk >> 6) & 1;
    bool more = (kk + 64) < D_;
    if (more) { issue(buf ^ 1, kk + 64); wait_async_le8(); }
    else      { wait_async_0(); }
    __syncthreads();
#pragma unroll
    for (int k2 = 0; k2 < 64; k2 += 32) {
      v16h af[2], bf[4];
#pragma unroll
      for (int i = 0; i < 2; ++i) af[i] = load_a_frag(&As[buf][wr + 16 * i][k2], 72, lane);
#pragma unroll
      for (int j = 0; j < 4; ++j) bf[j] = load_b_frag(&Bs[buf][wc + 16 * j][k2], 72, lane);
#pragma unroll
      for (int i = 0; i < 2; ++i)
#pragma unroll
        for (int j = 0; j < 4; ++j) acc[i][j] = wmma_f16(af[i], bf[j], acc[i][j]);
    }
    __syncthreads();
  }
#pragma unroll
  for (int i = 0; i < 2; ++i)
#pragma unroll
    for (int j = 0; j < 4; ++j) {
      int ncol  = nb + wc + 16 * j + (lane & 15);
      float bv  = bias[ncol];
      int mbase = mb + wr + 16 * i + ((lane & 16) ? 8 : 0);
#pragma unroll
      for (int r = 0; r < 8; ++r)
        Out[(size_t)(mbase + r) * D_ + ncol] = acc[i][j][r] + bv;
    }
}

// ------------------------------------------------- flash attention core
// grid (S/64, B*H), 128 threads = 4 waves; wave w owns query rows 16w..16w+15.
__global__ __launch_bounds__(128) void attn_kernel(
    const half_t* __restrict__ Qh, const half_t* __restrict__ Kh,
    const half_t* __restrict__ Vh, const unsigned char* __restrict__ mask,
    half_t* __restrict__ Xp) {
  __shared__ __align__(16) half_t Qs[64][72];
  __shared__ __align__(16) half_t Ks[32][72];
  __shared__ __align__(16) half_t Vt[64][40];      // V transposed: [dk][key]
  __shared__ __align__(16) half_t Ps[4][16][40];   // per-wave P scratch
  int tid = threadIdx.x, lane = tid & 31, w = tid >> 5;
  int bh = blockIdx.y, b = bh >> 4, h = bh & 15;
  int qbase = blockIdx.x * 64;
  const half_t* Qp = Qh + ((size_t)bh * S_ + qbase) * DK_;
#pragma unroll
  for (int i = 0; i < 4; ++i) {                    // Q tile 64x64
    int slot = tid + 128 * i;
    int r = slot >> 3, c = (slot & 7) << 3;
    *(v8h*)&Qs[r][c] = *(const v8h*)(Qp + r * DK_ + c);
  }
  const unsigned char* mp = mask + (size_t)b * S_;
  float rmax[8], rsum[8];
#pragma unroll
  for (int r = 0; r < 8; ++r) { rmax[r] = -3.0e38f; rsum[r] = 0.f; }
  v8f O[4] = {};
  __syncthreads();

  for (int kb = 0; kb < S_; kb += 32) {
    const half_t* Kp = Kh + ((size_t)bh * S_ + kb) * DK_;
    const half_t* Vp = Vh + ((size_t)bh * S_ + kb) * DK_;
#pragma unroll
    for (int i = 0; i < 2; ++i) {                  // K block + V transposed
      int slot = tid + 128 * i;
      int r = slot >> 3, c = (slot & 7) << 3;
      *(v8h*)&Ks[r][c] = *(const v8h*)(Kp + r * DK_ + c);
      v8h v = *(const v8h*)(Vp + r * DK_ + c);
#pragma unroll
      for (int jj = 0; jj < 8; ++jj) Vt[c + jj][r] = v[jj];
    }
    __syncthreads();
    // scores: 16 q rows x 32 keys
    v8f sc[2] = {};
    v16h aq0 = load_a_frag(&Qs[16 * w][0],  72, lane);
    v16h aq1 = load_a_frag(&Qs[16 * w][32], 72, lane);
#pragma unroll
    for (int t = 0; t < 2; ++t) {
      v16h bk0 = load_b_frag(&Ks[16 * t][0],  72, lane);
      v16h bk1 = load_b_frag(&Ks[16 * t][32], 72, lane);
      sc[t] = wmma_f16(aq0, bk0, sc[t]);
      sc[t] = wmma_f16(aq1, bk1, sc[t]);
    }
    // scale 1/sqrt(64) + mask (column = key)
#pragma unroll
    for (int t = 0; t < 2; ++t) {
      int key = kb + t * 16 + (lane & 15);
      bool keep = mp[key] != 0;
#pragma unroll
      for (int r = 0; r < 8; ++r)
        sc[t][r] = keep ? sc[t][r] * 0.125f : -1.0e9f;
    }
    // online softmax; row r lives on 16 lanes (halves of the wave)
    float nf[8];
#pragma unroll
    for (int r = 0; r < 8; ++r) {
      float mx = fmaxf(sc[0][r], sc[1][r]);
#pragma unroll
      for (int off = 8; off >= 1; off >>= 1)
        mx = fmaxf(mx, __shfl_xor(mx, off, 32));
      float nm = fmaxf(rmax[r], mx);
      float f  = __expf(rmax[r] - nm);
      float p0 = __expf(sc[0][r] - nm);
      float p1 = __expf(sc[1][r] - nm);
      sc[0][r] = p0; sc[1][r] = p1;
      float ps = p0 + p1;
#pragma unroll
      for (int off = 8; off >= 1; off >>= 1)
        ps += __shfl_xor(ps, off, 32);
      rsum[r] = rsum[r] * f + ps;
      rmax[r] = nm;
      nf[r]   = f;
    }
#pragma unroll
    for (int j = 0; j < 4; ++j)
#pragma unroll
      for (int r = 0; r < 8; ++r) O[j][r] *= nf[r];
    // P (C layout) -> per-wave LDS -> A fragment
    int rowoff = (lane & 16) ? 8 : 0;
#pragma unroll
    for (int t = 0; t < 2; ++t) {
      int col = t * 16 + (lane & 15);
#pragma unroll
      for (int r = 0; r < 8; ++r)
        Ps[w][rowoff + r][col] = (half_t)sc[t][r];
    }
    v16h ap = load_a_frag(&Ps[w][0][0], 40, lane);
#pragma unroll
    for (int j = 0; j < 4; ++j) {
      v16h bv = load_b_frag(&Vt[16 * j][0], 40, lane);
      O[j] = wmma_f16(ap, bv, O[j]);
    }
    __syncthreads();
  }
  // normalize + write into transposed (B,S,H*DK) layout, f16
  int rowoff = (lane & 16) ? 8 : 0;
#pragma unroll
  for (int j = 0; j < 4; ++j) {
    int dk = 16 * j + (lane & 15);
#pragma unroll
    for (int r = 0; r < 8; ++r) {
      int sq = qbase + 16 * w + rowoff + r;
      float val = O[j][r] / rsum[r];
      Xp[((size_t)b * S_ + sq) * D_ + h * DK_ + dk] = (half_t)val;
    }
  }
}

// ---------------------------------------------------------------- launcher
extern "C" void kernel_launch(void* const* d_in, const int* in_sizes, int n_in,
                              void* d_out, int out_size, void* d_ws, size_t ws_size,
                              hipStream_t stream) {
  (void)in_sizes; (void)n_in; (void)out_size; (void)ws_size;
  const float* query = (const float*)d_in[0];
  const float* key   = (const float*)d_in[1];
  const float* value = (const float*)d_in[2];
  const unsigned char* mask = (const unsigned char*)d_in[3];
  const float* Wq = (const float*)d_in[4];
  const float* bq = (const float*)d_in[5];
  const float* Wk = (const float*)d_in[6];
  const float* bk = (const float*)d_in[7];
  const float* Wv = (const float*)d_in[8];
  const float* bv = (const float*)d_in[9];
  const float* Wo = (const float*)d_in[10];
  const float* bo = (const float*)d_in[11];
  float* out = (float*)d_out;

  half_t* ws = (half_t*)d_ws;
  const size_t WSZ = (size_t)D_ * D_;   // 1M elems per weight
  const size_t TSZ = (size_t)M_ * D_;   // 8M elems per activation tensor
  half_t* X16q = ws;                    // f16 copies of the inputs
  half_t* X16k = X16q + TSZ;
  half_t* X16v = X16k + TSZ;
  half_t* Wq16 = X16v + TSZ;
  half_t* Wk16 = Wq16 + WSZ;
  half_t* Wv16 = Wk16 + WSZ;
  half_t* Wo16 = Wv16 + WSZ;
  half_t* Qh   = Wo16 + WSZ;
  half_t* Kh   = Qh + TSZ;
  half_t* Vh   = Kh + TSZ;
  half_t* Xp   = Vh + TSZ;

  int wBlocks = (int)(WSZ / 4 / 256);
  int tBlocks = (int)(TSZ / 4 / 256);
  cvt_f32_f16<<<tBlocks, 256, 0, stream>>>(query, X16q, (int)TSZ);
  cvt_f32_f16<<<tBlocks, 256, 0, stream>>>(key,   X16k, (int)TSZ);
  cvt_f32_f16<<<tBlocks, 256, 0, stream>>>(value, X16v, (int)TSZ);
  cvt_f32_f16<<<wBlocks, 256, 0, stream>>>(Wq, Wq16, (int)WSZ);
  cvt_f32_f16<<<wBlocks, 256, 0, stream>>>(Wk, Wk16, (int)WSZ);
  cvt_f32_f16<<<wBlocks, 256, 0, stream>>>(Wv, Wv16, (int)WSZ);
  cvt_f32_f16<<<wBlocks, 256, 0, stream>>>(Wo, Wo16, (int)WSZ);

  dim3 gg(D_ / 128, M_ / 128);
  gemm_qkv_kernel<<<gg, 256, 0, stream>>>(X16q, Wq16, bq, Qh);
  gemm_qkv_kernel<<<gg, 256, 0, stream>>>(X16k, Wk16, bk, Kh);
  gemm_qkv_kernel<<<gg, 256, 0, stream>>>(X16v, Wv16, bv, Vh);

  dim3 ga(S_ / 64, B_ * H_);
  attn_kernel<<<ga, 128, 0, stream>>>(Qh, Kh, Vh, mask, Xp);

  gemm_out_kernel<<<gg, 256, 0, stream>>>(Xp, Wo16, bo, out);
}
